// NTXentLoss_44538810860051
// MI455X (gfx1250) — matrix operand order repaired and checked by reading, not compiled, not run on hardware
//
#include <hip/hip_runtime.h>
#include <hip/hip_bf16.h>
#include <math.h>

#define BATCH   2048
#define MEMROWS 65536
#define DIM     64
#define KHARD   153
#define NBINS   512
#define CAP     192
#define NWAVES  8
#define TPB     256
#define HFLOOR  2.0f
#define HSCALE  64.0f   // NBINS / (10 - 2)

typedef __attribute__((ext_vector_type(16))) __bf16 v16bf;
typedef __attribute__((ext_vector_type(8)))  __bf16 v8bf;
typedef __attribute__((ext_vector_type(8)))  float  v8f;

// A-fragment: lane holds K = base..base+15 contiguous (matches B)
static __device__ inline v16bf load_fragA(const __bf16* p) {
    v8bf lo = *(const v8bf*)(p);
    v8bf hi = *(const v8bf*)(p + 8);
    return __builtin_shufflevector(lo, hi, 0,1,2,3,4,5,6,7,8,9,10,11,12,13,14,15);
}
// B-fragment: lanes 0-15 hold K=0..15 contiguous, lanes 16-31 hold K=16..31
static __device__ inline v16bf load_fragB(const __bf16* p) {
    return *(const v16bf*)(p);
}

static __device__ inline unsigned ordf(float f) {
    unsigned u = __float_as_uint(f);
    return (u & 0x80000000u) ? ~u : (u | 0x80000000u);
}
static __device__ inline float unordf(unsigned u) {
    return (u & 0x80000000u) ? __uint_as_float(u ^ 0x80000000u) : __uint_as_float(~u);
}

// ---------- normalization kernels ----------
__global__ void norm_mem_kernel(const float* __restrict__ mem,
                                float* __restrict__ memN, __bf16* __restrict__ memBF) {
    int row = blockIdx.x, t = threadIdx.x;               // 32 threads, 2 elems/thread
    float2 v = *(const float2*)(mem + (size_t)row * DIM + t * 2);
    float ss = v.x * v.x + v.y * v.y;
#pragma unroll
    for (int m = 16; m >= 1; m >>= 1) ss += __shfl_xor(ss, m, 32);
    float inv = 1.0f / fmaxf(sqrtf(ss), 1e-12f);
    float2 o; o.x = v.x * inv; o.y = v.y * inv;
    *(float2*)(memN + (size_t)row * DIM + t * 2) = o;
    memBF[(size_t)row * DIM + t * 2]     = (__bf16)o.x;
    memBF[(size_t)row * DIM + t * 2 + 1] = (__bf16)o.y;
}

__global__ void norm_z_kernel(const float* __restrict__ zi, const float* __restrict__ zj,
                              float* __restrict__ ziN, float* __restrict__ zjN,
                              __bf16* __restrict__ ziBF, __bf16* __restrict__ zjBF,
                              float* __restrict__ pos, float* __restrict__ out) {
    int b = blockIdx.x, t = threadIdx.x;                 // 32 threads
    if (b == 0 && t == 0) out[0] = 0.0f;                 // zero accumulator (runs before main)
    float2 vi = *(const float2*)(zi + (size_t)b * DIM + t * 2);
    float2 vj = *(const float2*)(zj + (size_t)b * DIM + t * 2);
    float si = vi.x * vi.x + vi.y * vi.y;
    float sj = vj.x * vj.x + vj.y * vj.y;
#pragma unroll
    for (int m = 16; m >= 1; m >>= 1) { si += __shfl_xor(si, m, 32); sj += __shfl_xor(sj, m, 32); }
    float ii = 1.0f / fmaxf(sqrtf(si), 1e-12f);
    float ij = 1.0f / fmaxf(sqrtf(sj), 1e-12f);
    float2 ni; ni.x = vi.x * ii; ni.y = vi.y * ii;
    float2 nj; nj.x = vj.x * ij; nj.y = vj.y * ij;
    *(float2*)(ziN + (size_t)b * DIM + t * 2) = ni;
    *(float2*)(zjN + (size_t)b * DIM + t * 2) = nj;
    ziBF[(size_t)b * DIM + t * 2]     = (__bf16)ni.x;
    ziBF[(size_t)b * DIM + t * 2 + 1] = (__bf16)ni.y;
    zjBF[(size_t)b * DIM + t * 2]     = (__bf16)nj.x;
    zjBF[(size_t)b * DIM + t * 2 + 1] = (__bf16)nj.y;
    float d = ni.x * nj.x + ni.y * nj.y;
#pragma unroll
    for (int m = 16; m >= 1; m >>= 1) d += __shfl_xor(d, m, 32);
    if (t == 0) pos[b] = d * 10.0f;                      // already scaled by 1/T
}

// ---------- main kernel: WMMA similarity + streaming top-k logsumexp ----------
__global__ __launch_bounds__(TPB) void ntxent_main(
    const __bf16* __restrict__ ziBF, const __bf16* __restrict__ zjBF,
    const float* __restrict__ ziN,  const float* __restrict__ zjN,
    const __bf16* __restrict__ memBF, const float* __restrict__ memN,
    const float* __restrict__ pos, const int* __restrict__ rnd, int nRand,
    float* __restrict__ out)
{
    __shared__ unsigned hist[16 * NBINS];   // 32 KB
    __shared__ float    lst[16 * CAP];      // 12 KB
    __shared__ unsigned lstCnt[16];
    __shared__ float    zlds[16 * DIM];     // 4 KB (f32 rows for rand gather)
    __shared__ unsigned rowMaxU[16];
    __shared__ float    rowM[16];
    __shared__ int      rowBin[16];
    __shared__ int      rowNeed[16];
    __shared__ float    rowTop[16];
    __shared__ float    rowSum2[16];
    __shared__ float    rowRand[16];
    __shared__ float    tstar[16];
    __shared__ int      gtCnt[16];

    const int tid  = threadIdx.x;
    const int lane = tid & 31;
    const int wave = tid >> 5;
    const int ln   = lane & 15;
    const int half = lane >> 4;
    const int rb8  = half * 8;              // C/D rows: M = d + 8*half
    const int rowBase = blockIdx.x * 16;
    const int side = blockIdx.y;

    const __bf16* zBF = side ? zjBF : ziBF;
    const float*  zN  = side ? zjN  : ziN;

    for (int i = tid; i < 16 * NBINS; i += TPB) hist[i] = 0u;
    for (int i = tid; i < 16 * DIM;   i += TPB) zlds[i] = zN[(size_t)rowBase * DIM + i];
    if (tid < 16) {
        lstCnt[tid] = 0u; rowMaxU[tid] = 0u;
        rowTop[tid] = 0.f; rowSum2[tid] = 0.f; rowRand[tid] = 0.f;
    }
    __syncthreads();

    // A fragments (16 rows x K=64), resident for whole kernel
    const __bf16* arow = zBF + (size_t)(rowBase + ln) * DIM;
    const int abase = half * 16;
    v16bf a0 = load_fragA(arow + abase);
    v16bf a1 = load_fragA(arow + 32 + abase);
    const int bofs = half * 16;

    // ---- Pass 1: row max + histogram of high values (2 N-tiles/wave/iter for ILP) ----
    float vmax[8];
#pragma unroll
    for (int d = 0; d < 8; ++d) vmax[d] = -1.0e30f;

    auto procHist = [&](const v8f& acc) {
#pragma unroll
        for (int d = 0; d < 8; ++d) {
            float s = acc[d] * 10.0f;
            vmax[d] = fmaxf(vmax[d], s);
            if (s >= HFLOOR) {
                int b = (int)((s - HFLOOR) * HSCALE);
                b = b > (NBINS - 1) ? (NBINS - 1) : b;
                atomicAdd(&hist[(rb8 + d) * NBINS + b], 1u);
            }
        }
    };

    for (int n = wave * 32; n < MEMROWS; n += 32 * NWAVES) {
        const __bf16* brow0 = memBF + (size_t)(n + ln) * DIM;
        const __bf16* brow1 = memBF + (size_t)(n + 16 + ln) * DIM;
        v16bf b00 = load_fragB(brow0 + bofs);
        v16bf b01 = load_fragB(brow0 + 32 + bofs);
        v16bf b10 = load_fragB(brow1 + bofs);
        v16bf b11 = load_fragB(brow1 + 32 + bofs);
        v8f acc0 = {0.f, 0.f, 0.f, 0.f, 0.f, 0.f, 0.f, 0.f};
        v8f acc1 = {0.f, 0.f, 0.f, 0.f, 0.f, 0.f, 0.f, 0.f};
        acc0 = __builtin_amdgcn_wmma_f32_16x16x32_bf16(false, a0, false, b00, (short)0, acc0, false, false);
        acc1 = __builtin_amdgcn_wmma_f32_16x16x32_bf16(false, a0, false, b10, (short)0, acc1, false, false);
        acc0 = __builtin_amdgcn_wmma_f32_16x16x32_bf16(false, a1, false, b01, (short)0, acc0, false, false);
        acc1 = __builtin_amdgcn_wmma_f32_16x16x32_bf16(false, a1, false, b11, (short)0, acc1, false, false);
        procHist(acc0);
        procHist(acc1);
    }
#pragma unroll
    for (int d = 0; d < 8; ++d) atomicMax(&rowMaxU[rb8 + d], ordf(vmax[d]));
    __syncthreads();

    // ---- Pass 2: find threshold bin per row ----
    if (tid < 16) {
        float M = unordf(rowMaxU[tid]);
        float p = pos[rowBase + tid];
        rowM[tid] = fmaxf(M, p);
        int cum = 0, B = 0;
        for (int b = NBINS - 1; b >= 0; --b) {
            int c = (int)hist[tid * NBINS + b];
            if (cum + c >= KHARD) { B = b; break; }
            cum += c;
        }
        rowBin[tid]  = B;
        rowNeed[tid] = KHARD - cum;
    }
    __syncthreads();

    // ---- Pass 3: recompute sims; sum exp above threshold bin, collect boundary bin ----
    float Mreg[8]; int Breg[8]; float lsum[8];
#pragma unroll
    for (int d = 0; d < 8; ++d) { Mreg[d] = rowM[rb8 + d]; Breg[d] = rowBin[rb8 + d]; lsum[d] = 0.f; }

    auto procSum = [&](const v8f& acc) {
#pragma unroll
        for (int d = 0; d < 8; ++d) {
            float s = acc[d] * 10.0f;
            int b = (s < HFLOOR) ? 0 : (int)((s - HFLOOR) * HSCALE);
            b = b > (NBINS - 1) ? (NBINS - 1) : b;
            if (b > Breg[d]) {
                lsum[d] += __expf(s - Mreg[d]);
            } else if (b == Breg[d]) {
                unsigned c = atomicAdd(&lstCnt[rb8 + d], 1u);
                if (c < CAP) lst[(rb8 + d) * CAP + c] = s;
            }
        }
    };

    for (int n = wave * 32; n < MEMROWS; n += 32 * NWAVES) {
        const __bf16* brow0 = memBF + (size_t)(n + ln) * DIM;
        const __bf16* brow1 = memBF + (size_t)(n + 16 + ln) * DIM;
        v16bf b00 = load_fragB(brow0 + bofs);
        v16bf b01 = load_fragB(brow0 + 32 + bofs);
        v16bf b10 = load_fragB(brow1 + bofs);
        v16bf b11 = load_fragB(brow1 + 32 + bofs);
        v8f acc0 = {0.f, 0.f, 0.f, 0.f, 0.f, 0.f, 0.f, 0.f};
        v8f acc1 = {0.f, 0.f, 0.f, 0.f, 0.f, 0.f, 0.f, 0.f};
        acc0 = __builtin_amdgcn_wmma_f32_16x16x32_bf16(false, a0, false, b00, (short)0, acc0, false, false);
        acc1 = __builtin_amdgcn_wmma_f32_16x16x32_bf16(false, a0, false, b10, (short)0, acc1, false, false);
        acc0 = __builtin_amdgcn_wmma_f32_16x16x32_bf16(false, a1, false, b01, (short)0, acc0, false, false);
        acc1 = __builtin_amdgcn_wmma_f32_16x16x32_bf16(false, a1, false, b11, (short)0, acc1, false, false);
        procSum(acc0);
        procSum(acc1);
    }
#pragma unroll
    for (int d = 0; d < 8; ++d) atomicAdd(&rowTop[rb8 + d], lsum[d]);
    __syncthreads();

    // ---- Pass 4: resolve boundary bin exactly (rank-count small LDS list) ----
    for (int r = 0; r < 16; ++r) {
        __syncthreads();
        int cnt  = (int)lstCnt[r];
        int n    = cnt < CAP ? cnt : CAP;
        int need = rowNeed[r];
        float M  = rowM[r];
        float* L = &lst[r * CAP];
        if (cnt > CAP || n <= need) {           // overflow fallback / take-all (uniform branch)
            float part = 0.f;
            for (int i = tid; i < n; i += TPB) part += __expf(L[i] - M);
            if (cnt > CAP && n > 0) part *= (float)need / (float)n;
            if (part != 0.f) atomicAdd(&rowSum2[r], part);
            continue;
        }
        for (int i = tid; i < n; i += TPB) {    // find need-th largest exactly
            float v = L[i];
            int gt = 0, eq = 0;
            for (int j = 0; j < n; ++j) { float u = L[j]; gt += (u > v); eq += (u == v); }
            if (gt < need && gt + eq >= need) { tstar[r] = v; gtCnt[r] = gt; }
        }
        __syncthreads();
        float tv = tstar[r];
        int  gtc = gtCnt[r];
        float part = 0.f;
        for (int i = tid; i < n; i += TPB) { float v = L[i]; if (v > tv) part += __expf(v - M); }
        if (tid == 0) part += (float)(need - gtc) * __expf(tv - M);
        if (part != 0.f) atomicAdd(&rowSum2[r], part);
    }
    __syncthreads();

    // ---- Pass 5: rand-negative gather (f32 path, matches reference sims) ----
    const int nPairs = 16 * nRand;
    for (int p = tid; p < nPairs; p += TPB) {
        int r = p / nRand;
        int j = p - r * nRand;
        int idx = rnd[j];
        const float4* mrow = (const float4*)(memN + (size_t)idx * DIM);
        const float* zr = &zlds[r * DIM];
        float dot = 0.f;
#pragma unroll
        for (int k = 0; k < DIM / 4; ++k) {
            float4 m4 = mrow[k];
            dot += zr[4 * k] * m4.x + zr[4 * k + 1] * m4.y + zr[4 * k + 2] * m4.z + zr[4 * k + 3] * m4.w;
        }
        atomicAdd(&rowRand[r], __expf(dot * 10.0f - rowM[r]));
    }
    __syncthreads();

    // ---- Finish: per-row loss, global mean ----
    if (tid < 16) {
        int r = tid;
        float p = pos[rowBase + r];
        float M = rowM[r];
        float total = rowTop[r] + rowSum2[r] + rowRand[r] + __expf(p - M);
        float loss = M + logf(total) - p;
        atomicAdd(out, loss * (1.0f / 4096.0f));
    }
}

extern "C" void kernel_launch(void* const* d_in, const int* in_sizes, int n_in,
                              void* d_out, int out_size, void* d_ws, size_t ws_size,
                              hipStream_t stream) {
    (void)n_in; (void)out_size; (void)ws_size;
    const float* zi  = (const float*)d_in[0];
    const float* zj  = (const float*)d_in[1];
    const float* mem = (const float*)d_in[2];
    const int*   rnd = (const int*)d_in[3];
    const int nRand  = in_sizes[3];

    char* ws = (char*)d_ws;
    float*  memNp = (float*) (ws);                        // 65536*64*4 = 16 MB
    __bf16* memBF = (__bf16*)(ws + 16777216);             //  8 MB
    float*  ziN   = (float*) (ws + 25165824);             // 512 KB
    float*  zjN   = (float*) (ws + 25690112);             // 512 KB
    __bf16* ziBF  = (__bf16*)(ws + 26214400);             // 256 KB
    __bf16* zjBF  = (__bf16*)(ws + 26476544);             // 256 KB
    float*  posb  = (float*) (ws + 26738688);             //   8 KB
    float*  outp  = (float*)d_out;

    norm_mem_kernel<<<MEMROWS, 32, 0, stream>>>(mem, memNp, memBF);
    norm_z_kernel<<<BATCH, 32, 0, stream>>>(zi, zj, ziN, zjN, ziBF, zjBF, posb, outp);
    dim3 grid(BATCH / 16, 2);
    ntxent_main<<<grid, TPB, 0, stream>>>(ziBF, zjBF, ziN, zjN, memBF, memNp, posb, rnd, nRand, outp);
}
